// NGCF_90134183674371
// MI455X (gfx1250) — matrix operand order repaired and checked by reading, not compiled, not run on hardware
//
#include <hip/hip_runtime.h>
#include <hip/hip_bf16.h>

// ---------------------------------------------------------------------------
// NGCF propagation for MI455X (gfx1250, wave32).
//   d_out[N,192]: cols 0-63 = ego0, 64-127 = ego1, 128-191 = ego2.
//   ego for layer li is read/written in place in d_out (stride 192).
//   d_ws: side[N*64] f32 (38.4 MB) followed by 4x 64x64 f16 weights (32 KB).
// ---------------------------------------------------------------------------

#define NUM_USERS 100000
#define NUM_ITEMS 50000
#define N_NODES   (NUM_USERS + NUM_ITEMS)   // 150000
#define EMB       64
#define OUT_COLS  192
#define N_TILES   (N_NODES / 16)            // 9375
#define SIDE_FLOATS ((size_t)N_NODES * EMB) // 9.6M
#define NEG_SLOPE 0.01f

typedef __attribute__((ext_vector_type(16))) _Float16 v16h;
typedef __attribute__((ext_vector_type(8)))  float    v8f;

__device__ __forceinline__ v8f wmma_f16(v16h a, v16h b, v8f c) {
  // (neg_a, A, neg_b, B, c_mod, C, reuse_a, reuse_b)
  return __builtin_amdgcn_wmma_f32_16x16x32_f16(false, a, false, b,
                                                (short)0, c, false, false);
}

// ---------------------------------------------------------------------------
// ego0 = concat(user_emb[user_idx], item_emb[item_idx]) -> d_out cols [0,64)
// ---------------------------------------------------------------------------
__global__ __launch_bounds__(256) void k_init_ego(
    const int* __restrict__ uidx, const int* __restrict__ iidx,
    const float* __restrict__ uemb, const float* __restrict__ iemb,
    float* __restrict__ out) {
  int idx = blockIdx.x * blockDim.x + threadIdx.x;  // N_NODES*EMB threads
  if (idx >= N_NODES * EMB) return;
  int n = idx >> 6;
  int d = idx & 63;
  float v;
  if (n < NUM_USERS) v = uemb[(size_t)uidx[n] * EMB + d];
  else               v = iemb[(size_t)iidx[n - NUM_USERS] * EMB + d];
  out[(size_t)n * OUT_COLS + d] = v;
}

// ---------------------------------------------------------------------------
// Convert the four 64x64 f32 weight matrices to row-major f16.
// Layout in o: [0]=Wg0 [1]=Wb0 [2]=Wg1 [3]=Wb1, each 4096 halves.
// ---------------------------------------------------------------------------
__global__ __launch_bounds__(256) void k_cvt_weights(
    const float* __restrict__ wg0, const float* __restrict__ wb0,
    const float* __restrict__ wg1, const float* __restrict__ wb1,
    _Float16* __restrict__ o) {
  int i = blockIdx.x * blockDim.x + threadIdx.x;
  if (i >= EMB * EMB) return;
  o[i]                 = (_Float16)wg0[i];
  o[EMB * EMB + i]     = (_Float16)wb0[i];
  o[2 * EMB * EMB + i] = (_Float16)wg1[i];
  o[3 * EMB * EMB + i] = (_Float16)wb1[i];
}

// ---------------------------------------------------------------------------
// side = 0
// ---------------------------------------------------------------------------
__global__ __launch_bounds__(256) void k_zero(float4* __restrict__ p, int n4) {
  int i = blockIdx.x * blockDim.x + threadIdx.x;
  if (i < n4) p[i] = make_float4(0.f, 0.f, 0.f, 0.f);
}

// ---------------------------------------------------------------------------
// SpMM: side[row] += val * ego[col].  One wave per edge; each lane carries a
// float2 of the 256B row (coalesced gather) + 2 global_atomic_add_f32 into
// the L2-resident side buffer.
// ---------------------------------------------------------------------------
__global__ __launch_bounds__(256) void k_spmm(
    const int* __restrict__ rows, const int* __restrict__ cols,
    const float* __restrict__ vals,
    const float* __restrict__ ego,  // stride OUT_COLS floats per node
    float* __restrict__ side, int n_edges) {
  int lane   = threadIdx.x & 31;
  int warp   = (blockIdx.x * blockDim.x + threadIdx.x) >> 5;
  int nwarps = (gridDim.x * blockDim.x) >> 5;
  for (int e = warp; e < n_edges; e += nwarps) {
    int   r = rows[e];
    int   c = cols[e];
    float v = vals[e];
    const float2* src = reinterpret_cast<const float2*>(ego + (size_t)c * OUT_COLS);
    float2 x = src[lane];
    float* dst = side + (size_t)r * EMB + lane * 2;
    unsafeAtomicAdd(dst,     v * x.x);
    unsafeAtomicAdd(dst + 1, v * x.y);
  }
}

// ---------------------------------------------------------------------------
// Dense phase (one wave per 16-node tile), all matrix math on WMMA:
//   accS = side16 @ Wg^T + bg      (2x K-chunk WMMA per 16-col n-tile)
//   accB = (ego16*side16) @ Wb^T + bb
//   out  = leaky(accS) + leaky(accB)   -> d_out cols of the next layer
//
// A operand (16x32 f16): lane m=L&15 holds row m; element j holds
//   K = kc*32 + hi*16 + j                   (hi = L>>4)   [contiguous order, matched in B]
// B operand (32x16 f16): lane n=L&15 holds col n; element j holds
//   K = kc*32 + hi*16 + j                                  [same order as A]
// C/D (16x16 f32): lane col n=L&15, VGPR r -> row m = r + hi*8.
// ---------------------------------------------------------------------------
__global__ __launch_bounds__(256) void k_dense(
    const float* __restrict__ side,          // [N,64]
    const float* __restrict__ ego,           // d_out + 64*li (stride 192)
    float* __restrict__ out,                 // d_out + 64*(li+1) (stride 192)
    const _Float16* __restrict__ wg16,       // [64][64]
    const _Float16* __restrict__ wb16,       // [64][64]
    const float* __restrict__ bg,
    const float* __restrict__ bb) {
  int wave = (blockIdx.x * blockDim.x + threadIdx.x) >> 5;
  if (wave >= N_TILES) return;               // wave-uniform: EXEC stays all-1s
  int lane = threadIdx.x & 31;
  int m    = lane & 15;
  int hi   = lane >> 4;
  int base = wave * 16;

  const float* srow = side + (size_t)(base + m) * EMB;
  const float* erow = ego  + (size_t)(base + m) * OUT_COLS;

  // Pack A operands for both K-chunks (side, and ego*side for the bi term).
  v16h aS[2], aB[2];
#pragma unroll
  for (int kc = 0; kc < 2; ++kc) {
#pragma unroll
    for (int j = 0; j < 16; ++j) {
      int k = kc * 32 + hi * 16 + j;
      float s = srow[k];
      float e = erow[k];
      aS[kc][j] = (_Float16)s;
      aB[kc][j] = (_Float16)(s * e);
    }
  }

#pragma unroll
  for (int nt = 0; nt < 4; ++nt) {
    int n = nt * 16 + m;                     // output column owned by lane
    const _Float16* wgr = wg16 + (size_t)n * EMB;
    const _Float16* wbr = wb16 + (size_t)n * EMB;
    v16h bS0 = *(const v16h*)(wgr + hi * 16);
    v16h bS1 = *(const v16h*)(wgr + 32 + hi * 16);
    v16h bB0 = *(const v16h*)(wbr + hi * 16);
    v16h bB1 = *(const v16h*)(wbr + 32 + hi * 16);

    float gb = bg[n], bbias = bb[n];
    v8f accS = {gb, gb, gb, gb, gb, gb, gb, gb};
    v8f accB = {bbias, bbias, bbias, bbias, bbias, bbias, bbias, bbias};

    accS = wmma_f16(aS[0], bS0, accS);
    accS = wmma_f16(aS[1], bS1, accS);
    accB = wmma_f16(aB[0], bB0, accB);
    accB = wmma_f16(aB[1], bB1, accB);

#pragma unroll
    for (int r = 0; r < 8; ++r) {
      float x = accS[r]; x = (x > 0.f) ? x : NEG_SLOPE * x;
      float y = accB[r]; y = (y > 0.f) ? y : NEG_SLOPE * y;
      out[(size_t)(base + r + hi * 8) * OUT_COLS + nt * 16 + m] = x + y;
    }
  }
}

// ---------------------------------------------------------------------------
extern "C" void kernel_launch(void* const* d_in, const int* in_sizes, int n_in,
                              void* d_out, int out_size, void* d_ws, size_t ws_size,
                              hipStream_t stream) {
  const int*   uidx = (const int*)d_in[0];
  const int*   iidx = (const int*)d_in[1];
  const int*   rows = (const int*)d_in[2];
  const int*   cols = (const int*)d_in[3];
  const float* vals = (const float*)d_in[4];
  const float* uemb = (const float*)d_in[5];
  const float* iemb = (const float*)d_in[6];
  const float* Wg0  = (const float*)d_in[7];
  const float* bg0  = (const float*)d_in[8];
  const float* Wb0  = (const float*)d_in[9];
  const float* bb0  = (const float*)d_in[10];
  const float* Wg1  = (const float*)d_in[11];
  const float* bg1  = (const float*)d_in[12];
  const float* Wb1  = (const float*)d_in[13];
  const float* bb1  = (const float*)d_in[14];
  const int n_edges = in_sizes[2];

  float*    out  = (float*)d_out;
  float*    side = (float*)d_ws;                       // 38.4 MB
  _Float16* w16  = (_Float16*)((char*)d_ws + SIDE_FLOATS * sizeof(float));

  // ego0 into d_out cols [0,64)
  {
    int total  = N_NODES * EMB;
    int blocks = (total + 255) / 256;
    k_init_ego<<<blocks, 256, 0, stream>>>(uidx, iidx, uemb, iemb, out);
  }
  // f16 weights
  k_cvt_weights<<<(EMB * EMB + 255) / 256, 256, 0, stream>>>(Wg0, Wb0, Wg1, Wb1, w16);

  const int n4          = (int)(SIDE_FLOATS / 4);
  const int zero_blocks = (n4 + 255) / 256;
  const int spmm_blocks = 6144;                        // grid-stride waves
  const int dense_blocks = (N_TILES + 7) / 8;          // 8 waves / block

  for (int li = 0; li < 2; ++li) {
    const float*    ego = out + 64 * li;               // stride OUT_COLS
    float*          nxt = out + 64 * (li + 1);
    const _Float16* wg  = w16 + (size_t)(2 * li) * EMB * EMB;
    const _Float16* wb  = w16 + (size_t)(2 * li + 1) * EMB * EMB;
    const float*    bg  = (li == 0) ? bg0 : bg1;
    const float*    bb  = (li == 0) ? bb0 : bb1;

    k_zero<<<zero_blocks, 256, 0, stream>>>((float4*)side, n4);
    k_spmm<<<spmm_blocks, 256, 0, stream>>>(rows, cols, vals, ego, side, n_edges);
    k_dense<<<dense_blocks, 256, 0, stream>>>(side, ego, nxt, wg, wb, bg, bb);
  }
}
